// DynamicFilterWithImageInput_79448305041861
// MI455X (gfx1250) — matrix operand order repaired and checked by reading, not compiled, not run on hardware
//
#include <hip/hip_runtime.h>
#include <hip/hip_bf16.h>

typedef __attribute__((ext_vector_type(16))) _Float16 v16h;
typedef __attribute__((ext_vector_type(8)))  float    v8f;

#define EPSV 1e-5f
#define B_   16
#define C_   256
#define H_   128
#define W_   128
#define KK_  25
#define CK2_ 6400
#define HW_  (H_*W_)

__device__ __forceinline__ float wave_sum(float v) {
    #pragma unroll
    for (int off = 16; off > 0; off >>= 1) v += __shfl_xor(v, off, 32);
    return v;
}
__device__ __forceinline__ float wave_max(float v) {
    #pragma unroll
    for (int off = 16; off > 0; off >>= 1) v = fmaxf(v, __shfl_xor(v, off, 32));
    return v;
}

// ---------------- Kernel 1: Conv2d(3,64,3,pad=1)+bias+BN+ReLU+global-avg-pool (partial sums)
__global__ void __launch_bounds__(256)
conv1_pool_kernel(const float* __restrict__ img, const float* __restrict__ wconv,
                  const float* __restrict__ bias,
                  const float* __restrict__ g1, const float* __restrict__ beta1,
                  const float* __restrict__ m1, const float* __restrict__ v1,
                  float* __restrict__ pool) {
    __shared__ float s_sum[64];
    const int tid  = threadIdx.x;
    const int b    = blockIdx.x >> 6;      // 64 tiles of 256 pixels per image
    const int tile = blockIdx.x & 63;
    const int p = tile * 256 + tid;
    const int h = p >> 7, w = p & 127;
    if (tid < 64) s_sum[tid] = 0.f;
    __syncthreads();

    float patch[27];
    #pragma unroll
    for (int ci = 0; ci < 3; ci++) {
        const float* base = img + (b*3 + ci) * HW_;
        #pragma unroll
        for (int dy = 0; dy < 3; dy++) {
            int hh = h + dy - 1;
            #pragma unroll
            for (int dx = 0; dx < 3; dx++) {
                int ww = w + dx - 1;
                float xv = 0.f;
                if (hh >= 0 && hh < H_ && ww >= 0 && ww < W_) xv = base[hh*W_ + ww];
                patch[ci*9 + dy*3 + dx] = xv;
            }
        }
    }
    const int lane = tid & 31;
    for (int c = 0; c < 64; c++) {
        float acc = bias[c];
        const float* wc = wconv + c*27;          // uniform -> scalar loads
        #pragma unroll
        for (int t = 0; t < 27; t++) acc += patch[t] * wc[t];
        float s   = g1[c] * rsqrtf(v1[c] + EPSV);
        float val = fmaxf((acc - m1[c]) * s + beta1[c], 0.f);
        val = wave_sum(val);
        if (lane == 0) atomicAdd(&s_sum[c], val);
    }
    __syncthreads();
    if (tid < 64) atomicAdd(&pool[b*64 + tid], s_sum[tid]);
}

// ---------------- Kernel 2: dense (16x64)@(64x6400) + bias + BN via v_wmma_f32_16x16x32_f16
__global__ void __launch_bounds__(128)
dense_wmma_kernel(const float* __restrict__ pool, const float* __restrict__ wfilt,
                  const float* __restrict__ bfilt,
                  const float* __restrict__ g2, const float* __restrict__ beta2,
                  const float* __restrict__ m2, const float* __restrict__ v2,
                  float* __restrict__ dense) {
    const float inv_hw = 1.0f / (float)HW_;
    const int lane = threadIdx.x & 31;
    const int wave = threadIdx.x >> 5;
    const int n0   = (blockIdx.x * 4 + wave) * 16;   // 400 tiles of N=16
    const int mlo  = lane & 15;
    const int hi   = (lane >> 4) & 1;
    const int ncol = n0 + mlo;

    v8f c = {};
    #pragma unroll
    for (int kk = 0; kk < 2; kk++) {
        v16h a, bm;
        #pragma unroll
        for (int r = 0; r < 8; r++) {
            // A (16x32 f16): lanes0-15 K in {0-7,16-23}, lanes16-31 K in {8-15,24-31}
            int g  = r >> 2, rr = r & 3;
            int ka = kk*32 + g*16 + hi*8 + rr*2;
            a[2*r]   = (_Float16)(pool[mlo*64 + ka    ] * inv_hw);
            a[2*r+1] = (_Float16)(pool[mlo*64 + ka + 1] * inv_hw);
            // B (32x16 f16): lanes0-15 K in {0-7,16-23}, lanes16-31 K in {8-15,24-31}; N = lane%16
            int kb = kk*32 + g*16 + hi*8 + rr*2;
            bm[2*r]   = (_Float16)wfilt[ncol*64 + kb    ];
            bm[2*r+1] = (_Float16)wfilt[ncol*64 + kb + 1];
        }
        c = __builtin_amdgcn_wmma_f32_16x16x32_f16(false, a, false, bm,
                                                   (short)0, c, false, false);
    }
    // epilogue: + bias, then eval-BN; C/D layout: VGPR r -> M = r + 8*hi, N = lane%16
    float s  = g2[ncol] * rsqrtf(v2[ncol] + EPSV);
    float bf = bfilt[ncol], mm = m2[ncol], bb = beta2[ncol];
    #pragma unroll
    for (int r = 0; r < 8; r++) {
        int m = r + hi*8;
        dense[m*CK2_ + ncol] = (c[r] + bf - mm) * s + bb;
    }
}

// ---------------- Kernel 3: softmax over C=256 for each (b, j), one wave per item
__global__ void __launch_bounds__(256)
softmax_kernel(const float* __restrict__ dense, float* __restrict__ filt) {
    const int wid  = blockIdx.x * 8 + (threadIdx.x >> 5);   // 400 waves
    const int lane = threadIdx.x & 31;
    const int b = wid / KK_;
    const int j = wid % KK_;
    const float* src = dense + b*CK2_ + j;
    float v[8];
    float mx = -1e30f;
    #pragma unroll
    for (int t = 0; t < 8; t++) {
        v[t] = src[(t*32 + lane) * KK_];
        mx = fmaxf(mx, v[t]);
    }
    mx = wave_max(mx);
    float sum = 0.f;
    #pragma unroll
    for (int t = 0; t < 8; t++) { v[t] = __expf(v[t] - mx); sum += v[t]; }
    sum = wave_sum(sum);
    const float inv = 1.0f / sum;
    float* dst = filt + b*CK2_ + j;
    #pragma unroll
    for (int t = 0; t < 8; t++) dst[(t*32 + lane) * KK_] = v[t] * inv;
}

// ---------------- Kernel 4: per-sample dynamic depthwise 5x5, reflect pad, LDS-tiled
__global__ void __launch_bounds__(256)
dwconv_kernel(const float* __restrict__ x, const float* __restrict__ filt,
              float* __restrict__ out) {
    __shared__ float tile[36][37];
    const int tid = threadIdx.x;
    const int t16 = blockIdx.x & 15;        // 4x4 tiles of 32x32
    const int bc  = blockIdx.x >> 4;        // b*256 + c
    const int h0  = (t16 >> 2) * 32;
    const int w0  = (t16 & 3)  * 32;
    const float* xin = x + bc * HW_;

    float f[25];
    const float* fp = filt + bc * KK_;      // bc*25 == b*6400 + c*25 (uniform -> scalar)
    #pragma unroll
    for (int i = 0; i < 25; i++) f[i] = fp[i];

    for (int idx = tid; idx < 36*36; idx += 256) {
        int ly = idx / 36, lx = idx % 36;
        int gh = h0 + ly - 2; gh = gh < 0 ? -gh : (gh >= H_ ? 2*H_ - 2 - gh : gh);
        int gw = w0 + lx - 2; gw = gw < 0 ? -gw : (gw >= W_ ? 2*W_ - 2 - gw : gw);
        tile[ly][lx] = xin[gh*W_ + gw];
    }
    __syncthreads();

    const int tx = tid & 31;
    const int ty = tid >> 5;                // 0..7, each thread does 4 rows
    float* op = out + bc*HW_ + (h0 + ty*4)*W_ + (w0 + tx);
    #pragma unroll
    for (int s = 0; s < 4; s++) {
        float acc = 0.f;
        #pragma unroll
        for (int i = 0; i < 5; i++)
            #pragma unroll
            for (int j = 0; j < 5; j++)
                acc += tile[ty*4 + s + i][tx + j] * f[i*5 + j];
        op[s*W_] = acc;
    }
}

extern "C" void kernel_launch(void* const* d_in, const int* in_sizes, int n_in,
                              void* d_out, int out_size, void* d_ws, size_t ws_size,
                              hipStream_t stream) {
    const float* x_feat  = (const float*)d_in[0];
    const float* raw_img = (const float*)d_in[1];
    const float* w_conv1 = (const float*)d_in[2];
    const float* b_conv1 = (const float*)d_in[3];
    const float* g1      = (const float*)d_in[4];
    const float* beta1   = (const float*)d_in[5];
    const float* m1      = (const float*)d_in[6];
    const float* v1      = (const float*)d_in[7];
    const float* w_filt  = (const float*)d_in[8];
    const float* b_filt  = (const float*)d_in[9];
    const float* g2      = (const float*)d_in[10];
    const float* beta2   = (const float*)d_in[11];
    const float* m2      = (const float*)d_in[12];
    const float* v2      = (const float*)d_in[13];
    float* out = (float*)d_out;

    float* pool  = (float*)d_ws;            // 16*64
    float* dense = pool + 1024;             // 16*6400 (post-BN dense)
    float* filtb = dense + B_*CK2_;         // 16*6400 (post-softmax filters)

    hipMemsetAsync(pool, 0, 1024 * sizeof(float), stream);
    conv1_pool_kernel<<<B_*64, 256, 0, stream>>>(raw_img, w_conv1, b_conv1,
                                                 g1, beta1, m1, v1, pool);
    dense_wmma_kernel<<<100, 128, 0, stream>>>(pool, w_filt, b_filt,
                                               g2, beta2, m2, v2, dense);
    softmax_kernel<<<50, 256, 0, stream>>>(dense, filtb);
    dwconv_kernel<<<B_*C_*16, 256, 0, stream>>>(x_feat, filtb, out);
}
